// Rewirescorelayer_781684048167
// MI455X (gfx1250) — matrix operand order repaired and checked by reading, not compiled, not run on hardware
//
#include <hip/hip_runtime.h>
#include <hip/hip_fp16.h>

#define NGRAPH 16
#define WSZ    32
#define RNUM   4
#define HNUM   8
#define DHEAD  64
#define FINF   256
#define NNODES 4033
#define WP     132

typedef __attribute__((ext_vector_type(16))) _Float16 v16h;
typedef __attribute__((ext_vector_type(8)))  _Float16 v8h;
typedef __attribute__((ext_vector_type(8)))  float    v8f;
typedef __attribute__((ext_vector_type(4)))  float    v4f;

__constant__ int d_len[NGRAPH]     = {251,247,263,255,240,258,249,260,252,245,256,250,248,261,244,254};
__constant__ int d_nw[NGRAPH]      = {8,8,9,8,8,9,8,9,8,8,8,8,8,9,8,8};
__constant__ int d_twstart[NGRAPH] = {0,8,16,25,33,41,50,58,67,75,83,91,99,107,116,124};
__constant__ int d_nbase[NGRAPH]   = {0,251,498,761,1016,1256,1514,1763,2023,2275,2520,2776,3026,3274,3535,3779};

__device__ inline int win_graph(int wt) {
    int g = 0;
#pragma unroll
    for (int i = 1; i < NGRAPH; ++i) g += (wt >= d_twstart[i]) ? 1 : 0;
    return g;
}

// ---------------------------------------------------------------------------
// Pass 1: Q/K projection.  One block per (r, trimmed-window).  8 waves:
// waves 0..3 -> Q cols [wv*128,+128), waves 4..7 -> K cols.  Output layout
// f16 [r][wt][h][slot(32)][d(64)] so that attention A (Q rows) and B
// (K columns over d) loads are contiguous.
// ---------------------------------------------------------------------------
__global__ __launch_bounds__(256)
void proj_kernel(const float* __restrict__ X,
                 const float* __restrict__ Wq, const float* __restrict__ bq,
                 const float* __restrict__ Wk, const float* __restrict__ bk,
                 _Float16* __restrict__ Qs, _Float16* __restrict__ Ks)
{
    const int blk  = blockIdx.x;
    const int r    = blk / WP;
    const int wt   = blk % WP;
    const int g    = win_graph(wt);
    const int lw   = wt - d_twstart[g];
    const int vcnt = min(WSZ, d_len[g] - lw * WSZ);
    const int node0 = d_nbase[g] + lw * WSZ;

    const int lane = threadIdx.x;        // 0..31
    const int wv   = threadIdx.y;        // 0..7
    const bool doQ = (wv < 4);
    const float* W   = doQ ? Wq : Wk;
    const float* Bv  = doQ ? bq : bk;
    _Float16*    Out = doQ ? Qs : Ks;
    const int ncol0 = (wv & 3) * 128;

    const int ml = lane & 15;
    const int hi = lane >> 4;            // 0 or 1

    v8f acc[2][8];
    v8f z8 = {};
#pragma unroll
    for (int mt = 0; mt < 2; ++mt)
#pragma unroll
        for (int nt = 0; nt < 8; ++nt) acc[mt][nt] = z8;

#pragma unroll 1
    for (int k0 = 0; k0 < FINF; k0 += 32) {
        // A tiles: rows = window slots (f32 features -> f16)
        v16h a[2];
        v16h z16 = {};
#pragma unroll
        for (int mt = 0; mt < 2; ++mt) {
            int slot = mt * 16 + ml;
            if (slot < vcnt) {
                const float* xp = X + ((size_t)r * NNODES + (node0 + slot)) * FINF + k0 + hi * 8;
                const v4f* xp4 = (const v4f*)xp;
                v4f x0 = xp4[0], x1 = xp4[1];           // K = k0+c .. +8
                v4f x2 = xp4[4], x3 = xp4[5];           // K = k0+c+16 .. +8
                v16h av;
#pragma unroll
                for (int i = 0; i < 4; ++i) {
                    av[i]      = (_Float16)x0[i];
                    av[4 + i]  = (_Float16)x1[i];
                    av[8 + i]  = (_Float16)x2[i];
                    av[12 + i] = (_Float16)x3[i];
                }
                a[mt] = av;
            } else {
                a[mt] = z16;
            }
        }
#pragma unroll
        for (int nt = 0; nt < 8; ++nt) {
            int n = ncol0 + nt * 16 + ml;               // output column 0..511
            const v4f* wp4 = (const v4f*)(W + (size_t)n * FINF + k0 + hi * 8);
            v4f w0 = wp4[0], w1 = wp4[1], w2 = wp4[4], w3 = wp4[5];
            v16h bvec;
#pragma unroll
            for (int i = 0; i < 4; ++i) {
                bvec[i]      = (_Float16)w0[i];
                bvec[4 + i]  = (_Float16)w1[i];
                bvec[8 + i]  = (_Float16)w2[i];
                bvec[12 + i] = (_Float16)w3[i];
            }
#pragma unroll
            for (int mt = 0; mt < 2; ++mt)
                acc[mt][nt] = __builtin_amdgcn_wmma_f32_16x16x32_f16(
                    false, a[mt], false, bvec, (short)0, acc[mt][nt], false, false);
        }
    }

    // bias + store f16 scratch [r][wt][h][slot][d]
#pragma unroll
    for (int nt = 0; nt < 8; ++nt) {
        int n = ncol0 + nt * 16 + ml;
        float bb = Bv[n];
        int h = n >> 6;
        int d = n & 63;
#pragma unroll
        for (int mt = 0; mt < 2; ++mt) {
#pragma unroll
            for (int v = 0; v < 8; ++v) {
                int slot = mt * 16 + v + hi * 8;
                size_t idx = ((((size_t)r * WP + wt) * HNUM + h) * WSZ + slot) * DHEAD + d;
                Out[idx] = (_Float16)(acc[mt][nt][v] + bb);
            }
        }
    }
}

// ---------------------------------------------------------------------------
// Pass 2: windowed attention scores.  One block per (r, trimmed-window),
// wave = head.  sim[32x96] via 2x6 WMMA tiles, masked softmax over 96 keys,
// head-mean accumulated in LDS, banded coalesced scatter into zeroed output.
// ---------------------------------------------------------------------------
__global__ __launch_bounds__(256)
void attn_kernel(const _Float16* __restrict__ Qs, const _Float16* __restrict__ Ks,
                 float* __restrict__ out)
{
    __shared__ float score[WSZ][3 * WSZ];   // 12 KB

    const int blk  = blockIdx.x;
    const int r    = blk / WP;
    const int wt   = blk % WP;
    const int g    = win_graph(wt);
    const int lw   = wt - d_twstart[g];
    const int nwg  = d_nw[g];
    const int vcnt = min(WSZ, d_len[g] - lw * WSZ);
    const int node0 = d_nbase[g] + lw * WSZ;

    const int lane = threadIdx.x;
    const int h    = threadIdx.y;
    const int tid  = h * 32 + lane;
    const int ml   = lane & 15;
    const int hi   = lane >> 4;

    for (int e = tid; e < WSZ * 3 * WSZ; e += 256) (&score[0][0])[e] = 0.0f;
    __syncthreads();

    bool wok[3]; int kvc[3];
#pragma unroll
    for (int wo = 0; wo < 3; ++wo) {
        int lwk = lw + wo - 1;
        bool ok = (lwk >= 0) && (lwk < nwg);
        wok[wo] = ok;
        kvc[wo] = ok ? min(WSZ, d_len[g] - lwk * WSZ) : 0;
    }

    v8f acc[2][6];
    v8f z8 = {};
#pragma unroll
    for (int mt = 0; mt < 2; ++mt)
#pragma unroll
        for (int nt = 0; nt < 6; ++nt) acc[mt][nt] = z8;

#pragma unroll
    for (int kv = 0; kv < 2; ++kv) {
        int d0 = kv * 32;
        v16h a[2];
#pragma unroll
        for (int mt = 0; mt < 2; ++mt) {
            int q = mt * 16 + ml;
            const v8h* qp8 = (const v8h*)(Qs +
                ((((size_t)r * WP + wt) * HNUM + h) * WSZ + q) * DHEAD + d0 + hi * 16);
            v8h lo = qp8[0];
            v8h hc = qp8[1];                 // +8 halves
            a[mt] = __builtin_shufflevector(lo, hc,
                     0,1,2,3,4,5,6,7,8,9,10,11,12,13,14,15);
        }
#pragma unroll
        for (int nt = 0; nt < 6; ++nt) {
            int wo = nt >> 1;
            if (wok[wo]) {                  // wave-uniform branch: EXEC stays all-1s
                int wk  = wt + wo - 1;
                int key = (nt & 1) * 16 + ml;
                v16h bvec = *(const v16h*)(Ks +
                    ((((size_t)r * WP + wk) * HNUM + h) * WSZ + key) * DHEAD + d0 + hi * 16);
#pragma unroll
                for (int mt = 0; mt < 2; ++mt)
                    acc[mt][nt] = __builtin_amdgcn_wmma_f32_16x16x32_f16(
                        false, a[mt], false, bvec, (short)0, acc[mt][nt], false, false);
            }
        }
    }

    const float scale = 0.125f;             // D^-0.5, D=64
#pragma unroll
    for (int mt = 0; mt < 2; ++mt) {
#pragma unroll
        for (int v = 0; v < 8; ++v) {
            int q = mt * 16 + v + hi * 8;
            float x[6]; bool val[6];
#pragma unroll
            for (int nt = 0; nt < 6; ++nt) {
                int wo = nt >> 1;
                int kl = (nt & 1) * 16 + ml;
                val[nt] = wok[wo] && (kl < kvc[wo]);
                x[nt] = val[nt] ? acc[mt][nt][v] * scale : -__builtin_inff();
            }
            float mx = x[0];
#pragma unroll
            for (int nt = 1; nt < 6; ++nt) mx = fmaxf(mx, x[nt]);
#pragma unroll
            for (int m = 1; m <= 8; m <<= 1) mx = fmaxf(mx, __shfl_xor(mx, m, 32));
            float e[6]; float s = 0.0f;
#pragma unroll
            for (int nt = 0; nt < 6; ++nt) {
                e[nt] = val[nt] ? __expf(x[nt] - mx) : 0.0f;
                s += e[nt];
            }
#pragma unroll
            for (int m = 1; m <= 8; m <<= 1) s += __shfl_xor(s, m, 32);
            float inv = 0.125f / s;         // 1/H head mean folded in
            if (q < vcnt) {
#pragma unroll
                for (int nt = 0; nt < 6; ++nt)
                    if (val[nt]) atomicAdd(&score[q][nt * 16 + ml], e[nt] * inv);
            }
        }
    }
    __syncthreads();

    const size_t ldout = (size_t)RNUM * NNODES;
    for (int e2 = tid; e2 < WSZ * 3 * WSZ; e2 += 256) {
        int q  = e2 / (3 * WSZ);
        int j  = e2 % (3 * WSZ);
        int wo = j >> 5;
        int kl = j & 31;
        if (q < vcnt && wok[wo] && kl < kvc[wo]) {
            int knode = d_nbase[g] + (lw + wo - 1) * WSZ + kl;
            out[(size_t)(node0 + q) * ldout + (size_t)r * NNODES + knode] = score[q][j];
        }
    }
}

extern "C" void kernel_launch(void* const* d_in, const int* in_sizes, int n_in,
                              void* d_out, int out_size, void* d_ws, size_t ws_size,
                              hipStream_t stream) {
    (void)in_sizes; (void)n_in;
    const float* X  = (const float*)d_in[0];
    const float* Wq = (const float*)d_in[1];
    const float* bq = (const float*)d_in[2];
    const float* Wk = (const float*)d_in[3];
    const float* bk = (const float*)d_in[4];
    float* out = (float*)d_out;

    const size_t qk_elems = (size_t)RNUM * WP * HNUM * WSZ * DHEAD;  // 8,650,752 halves each
    if (ws_size < 2 * qk_elems * sizeof(_Float16)) return;
    _Float16* Qs = (_Float16*)d_ws;
    _Float16* Ks = Qs + qk_elems;

    // Output is >97% zeros (banded structure): zero it at HBM bandwidth.
    hipMemsetAsync(d_out, 0, (size_t)out_size * sizeof(float), stream);

    dim3 blkdim(32, 8);
    dim3 grid(RNUM * WP);
    proj_kernel<<<grid, blkdim, 0, stream>>>(X, Wq, bq, Wk, bk, Qs, Ks);
    attn_kernel<<<grid, blkdim, 0, stream>>>(Qs, Ks, out);
    (void)out_size;
}
